// TripletGCNModel_27015344292379
// MI455X (gfx1250) — matrix operand order repaired and checked by reading, not compiled, not run on hardware
//
#include <hip/hip_runtime.h>

// Problem constants (from the reference)
#define N_NODES 20000
#define N_EDGES 100000
#define DN 256
#define DE 256
#define DH 512
#define NLAYERS 2

// Tile config: 256 threads = 8 waves (wave32); 4x2 wave grid, 32x32 per wave.
#define BM 128
#define BN 64
#define BK 32
#define TPB 256
#define LDA 34   // padded LDS stride (elements) -> 17 dwords, conflict-free

typedef __attribute__((ext_vector_type(16))) __bf16 v16bf;
typedef __attribute__((ext_vector_type(8)))  float  v8f;

union FragAB { v16bf v; unsigned int u[8]; };
union Pack8  { uint4 v; unsigned int w[4]; unsigned short s[8]; };

__device__ __forceinline__ unsigned short f2bf(float f) {
  unsigned int u = __builtin_bit_cast(unsigned int, f);
  u += 0x7FFFu + ((u >> 16) & 1u);           // round-to-nearest-even
  return (unsigned short)(u >> 16);
}

// ---- LDS fragment loads matching CDNA5 ISA 7.12.2 16-bit layouts ----
// A 16x32: lanes 0-15 row=lane, K {0..7,16..23}; lanes 16-31 row=lane-16, K {8..15,24..31}
__device__ __forceinline__ void load_fragA(const unsigned short* lds, int rowBase, int lane, FragAB& f) {
  int row = rowBase + (lane & 15);
  int kb = (lane & 16) ? 8 : 0;
  const unsigned short* p = lds + row * LDA + kb;
#pragma unroll
  for (int i = 0; i < 4; ++i) {
    f.u[i]     = *(const unsigned int*)(p + 2 * i);
    f.u[i + 4] = *(const unsigned int*)(p + 16 + 2 * i);
  }
}
// B 32x16: lanes 0-15 col=lane, K {0..7,16..23}; lanes 16-31 col=lane-16, K {8..15,24..31} (Blds is [n][k])
__device__ __forceinline__ void load_fragB(const unsigned short* lds, int colBase, int lane, FragAB& f) {
  int n = colBase + (lane & 15);
  int kb = (lane & 16) ? 8 : 0;
  const unsigned int* p = (const unsigned int*)(lds + n * LDA + kb);
#pragma unroll
  for (int i = 0; i < 4; ++i) { f.u[i] = p[i]; f.u[i + 4] = p[8 + i]; }
}

// ---- register-staged tile movement (software pipelined) ----
// A: thread owns rows r0 = t>>2 (+64) with 8 contiguous k at kc = (t&3)*8 -> one b128 each.
__device__ __forceinline__ uint4 fetchA8(const unsigned short* p) { return *(const uint4*)p; }
__device__ __forceinline__ void commitA8(uint4 r, unsigned short* d) {
  Pack8 u; u.v = r;
#pragma unroll
  for (int i = 0; i < 4; ++i) *(unsigned int*)(d + 2 * i) = u.w[i];
}
// B: thread owns k-row k = t>>3, 8 contiguous n at n8 = (t&7)*8 -> one b128, transpose on LDS store.
__device__ __forceinline__ uint4 fetchB8(const unsigned short* __restrict__ W, int ldw, int kt, int n0, int t) {
  int k = t >> 3, n8 = (t & 7) * 8;
  return *(const uint4*)(W + (size_t)(kt + k) * ldw + n0 + n8);
}
__device__ __forceinline__ void commitB8(uint4 r, unsigned short* Blds, int t) {
  int k = t >> 3, n8 = (t & 7) * 8;
  Pack8 u; u.v = r;
#pragma unroll
  for (int i = 0; i < 8; ++i) Blds[(n8 + i) * LDA + k] = u.s[i];
}

#define WMMA_BF16(A, B, C) \
  __builtin_amdgcn_wmma_f32_16x16x32_bf16(false, (A), false, (B), (short)0, (C), false, false)

// ---------------- GEMM1: m1 = relu(gather(x,e) @ W1a + b1a) -> bf16 (E,512) --------------
// Rows beyond E are clamped (never stored); region select on uniform kt; reg double-buffered.
__global__ __launch_bounds__(TPB)
void gemm1_gather_kernel(const unsigned short* __restrict__ xb, const unsigned short* __restrict__ eb,
                         const int* __restrict__ srcIdx, const int* __restrict__ dstIdx,
                         const unsigned short* __restrict__ W, const float* __restrict__ bias,
                         unsigned short* __restrict__ out) {
  __shared__ unsigned short Alds[BM * LDA];
  __shared__ unsigned short Blds[BN * LDA];
  const int t = threadIdx.x, lane = t & 31, wave = t >> 5;
  const int wm = wave >> 1, wn = wave & 1;
  const int row0 = blockIdx.x * BM, n0 = blockIdx.y * BN;

  const int r0 = t >> 2;
  const int kc = (t & 3) * 8;
  const int g0 = min(row0 + r0, N_EDGES - 1);
  const int g1 = min(row0 + r0 + 64, N_EDGES - 1);
  const int di0 = dstIdx[g0], si0 = srcIdx[g0];
  const int di1 = dstIdx[g1], si1 = srcIdx[g1];
  const unsigned short* pd0 = xb + (size_t)di0 * DN + kc;
  const unsigned short* pd1 = xb + (size_t)di1 * DN + kc;
  const unsigned short* pe0 = eb + (size_t)g0 * DE + kc;
  const unsigned short* pe1 = eb + (size_t)g1 * DE + kc;
  const unsigned short* ps0 = xb + (size_t)si0 * DN + kc;
  const unsigned short* ps1 = xb + (size_t)si1 * DN + kc;
  unsigned short* dA0 = Alds + r0 * LDA + kc;
  unsigned short* dA1 = Alds + (r0 + 64) * LDA + kc;

  auto aptr = [&](int kt, const unsigned short*& q0, const unsigned short*& q1) {
    if (kt < DN)           { q0 = pd0 + kt;           q1 = pd1 + kt; }
    else if (kt < DN + DE) { q0 = pe0 + kt - DN;      q1 = pe1 + kt - DN; }
    else                   { q0 = ps0 + kt - DN - DE; q1 = ps1 + kt - DN - DE; }
  };

  v8f zero = {};
  v8f acc[2][2] = {{zero, zero}, {zero, zero}};

  const int NK = (2 * DN + DE) / BK;  // 24
  const unsigned short *q0, *q1;
  aptr(0, q0, q1);
  uint4 ra0 = fetchA8(q0), ra1 = fetchA8(q1);
  uint4 rb = fetchB8(W, DH, 0, n0, t);

  for (int i = 0; i < NK; ++i) {
    __syncthreads();
    commitA8(ra0, dA0);
    commitA8(ra1, dA1);
    commitB8(rb, Blds, t);
    __syncthreads();
    if (i + 1 < NK) {               // uniform branch: prefetch next tile into registers
      int kt = (i + 1) * BK;
      aptr(kt, q0, q1);
      ra0 = fetchA8(q0); ra1 = fetchA8(q1);
      rb = fetchB8(W, DH, kt, n0, t);
    }
    FragAB a[2], b[2];
    load_fragA(Alds, wm * 32, lane, a[0]);
    load_fragA(Alds, wm * 32 + 16, lane, a[1]);
    load_fragB(Blds, wn * 32, lane, b[0]);
    load_fragB(Blds, wn * 32 + 16, lane, b[1]);
#pragma unroll
    for (int fm = 0; fm < 2; ++fm)
#pragma unroll
      for (int fn = 0; fn < 2; ++fn) acc[fm][fn] = WMMA_BF16(a[fm].v, b[fn].v, acc[fm][fn]);
  }

#pragma unroll
  for (int fm = 0; fm < 2; ++fm)
#pragma unroll
    for (int fn = 0; fn < 2; ++fn)
#pragma unroll
      for (int v = 0; v < 8; ++v) {
        int row = row0 + wm * 32 + fm * 16 + v + ((lane & 16) ? 8 : 0);
        int col = n0 + wn * 32 + fn * 16 + (lane & 15);
        if (row < N_EDGES) out[(size_t)row * DH + col] = f2bf(fmaxf(acc[fm][fn][v] + bias[col], 0.f));
      }
}

// ----- GEMM2 fused: z = relu(m1 @ W1b + b1b); xm cols -> atomic segment sum; e cols -> e_out
__global__ __launch_bounds__(TPB)
void gemm2_fused_kernel(const unsigned short* __restrict__ m1, const unsigned short* __restrict__ W,
                        const float* __restrict__ bias, const int* __restrict__ dstIdx,
                        float* __restrict__ agg, float* __restrict__ e_out) {
  __shared__ unsigned short Alds[BM * LDA];
  __shared__ unsigned short Blds[2][BN * LDA];
  const int t = threadIdx.x, lane = t & 31, wave = t >> 5;
  const int wm = wave >> 1, wn = wave & 1;
  const int row0 = blockIdx.x * BM;
  const bool xm_mode = (blockIdx.y < 8);
  const int n0 = xm_mode ? blockIdx.y * BN : (DH + (blockIdx.y - 8) * BN);
  const int NLD = 2 * DH + DE;  // 1280

  const int r0 = t >> 2;
  const int kc = (t & 3) * 8;
  const int g0 = min(row0 + r0, N_EDGES - 1);
  const int g1 = min(row0 + r0 + 64, N_EDGES - 1);
  const unsigned short* pa0 = m1 + (size_t)g0 * DH + kc;
  const unsigned short* pa1 = m1 + (size_t)g1 * DH + kc;
  unsigned short* dA0 = Alds + r0 * LDA + kc;
  unsigned short* dA1 = Alds + (r0 + 64) * LDA + kc;

  v8f zero = {};
  v8f acc0[2][2] = {{zero, zero}, {zero, zero}};
  v8f acc1[2][2] = {{zero, zero}, {zero, zero}};

  const int NK = DH / BK;  // 16
  uint4 ra0 = fetchA8(pa0), ra1 = fetchA8(pa1);
  uint4 rb0 = fetchB8(W, NLD, 0, n0, t);
  uint4 rb1 = xm_mode ? fetchB8(W, NLD, 0, n0 + DH + DE, t) : rb0;

  for (int i = 0; i < NK; ++i) {
    __syncthreads();
    commitA8(ra0, dA0);
    commitA8(ra1, dA1);
    commitB8(rb0, Blds[0], t);
    if (xm_mode) commitB8(rb1, Blds[1], t);
    __syncthreads();
    if (i + 1 < NK) {
      int kt = (i + 1) * BK;
      ra0 = fetchA8(pa0 + kt); ra1 = fetchA8(pa1 + kt);
      rb0 = fetchB8(W, NLD, kt, n0, t);
      if (xm_mode) rb1 = fetchB8(W, NLD, kt, n0 + DH + DE, t);
    }
    FragAB a[2], b0[2], b1[2];
    load_fragA(Alds, wm * 32, lane, a[0]);
    load_fragA(Alds, wm * 32 + 16, lane, a[1]);
    load_fragB(Blds[0], wn * 32, lane, b0[0]);
    load_fragB(Blds[0], wn * 32 + 16, lane, b0[1]);
#pragma unroll
    for (int fm = 0; fm < 2; ++fm)
#pragma unroll
      for (int fn = 0; fn < 2; ++fn) acc0[fm][fn] = WMMA_BF16(a[fm].v, b0[fn].v, acc0[fm][fn]);
    if (xm_mode) {  // block-uniform branch; EXEC stays all-ones for WMMA
      load_fragB(Blds[1], wn * 32, lane, b1[0]);
      load_fragB(Blds[1], wn * 32 + 16, lane, b1[1]);
#pragma unroll
      for (int fm = 0; fm < 2; ++fm)
#pragma unroll
        for (int fn = 0; fn < 2; ++fn) acc1[fm][fn] = WMMA_BF16(a[fm].v, b1[fn].v, acc1[fm][fn]);
    }
  }

#pragma unroll
  for (int fm = 0; fm < 2; ++fm)
#pragma unroll
    for (int fn = 0; fn < 2; ++fn)
#pragma unroll
      for (int v = 0; v < 8; ++v) {
        int row = row0 + wm * 32 + fm * 16 + v + ((lane & 16) ? 8 : 0);
        int col = n0 + wn * 32 + fn * 16 + (lane & 15);
        if (row < N_EDGES) {
          if (xm_mode) {
            float v0 = fmaxf(acc0[fm][fn][v] + bias[col], 0.f);
            float v1 = fmaxf(acc1[fm][fn][v] + bias[col + DH + DE], 0.f);
            atomicAdd(&agg[(size_t)dstIdx[row] * DH + col], v0 + v1);
          } else {
            e_out[(size_t)row * DE + (col - DH)] = fmaxf(acc0[fm][fn][v] + bias[col], 0.f);
          }
        }
      }
}

// ----- Plain GEMM: BF16OUT: out=relu(A@W+b) bf16 ; else: out=resid+A@W+b (opt relu) f32 -----
template <bool BF16OUT>
__global__ __launch_bounds__(TPB)
void gemm_plain_kernel(const unsigned short* __restrict__ A, int M, int K, int Ncols,
                       const unsigned short* __restrict__ W, const float* __restrict__ bias,
                       unsigned short* __restrict__ outBf, float* __restrict__ outF,
                       const float* __restrict__ resid, int do_relu) {
  __shared__ unsigned short Alds[BM * LDA];
  __shared__ unsigned short Blds[BN * LDA];
  const int t = threadIdx.x, lane = t & 31, wave = t >> 5;
  const int wm = wave >> 1, wn = wave & 1;
  const int row0 = blockIdx.x * BM, n0 = blockIdx.y * BN;

  const int r0 = t >> 2;
  const int kc = (t & 3) * 8;
  const int g0 = min(row0 + r0, M - 1);
  const int g1 = min(row0 + r0 + 64, M - 1);
  const unsigned short* pa0 = A + (size_t)g0 * K + kc;
  const unsigned short* pa1 = A + (size_t)g1 * K + kc;
  unsigned short* dA0 = Alds + r0 * LDA + kc;
  unsigned short* dA1 = Alds + (r0 + 64) * LDA + kc;

  v8f zero = {};
  v8f acc[2][2] = {{zero, zero}, {zero, zero}};

  const int NK = K / BK;
  uint4 ra0 = fetchA8(pa0), ra1 = fetchA8(pa1);
  uint4 rb = fetchB8(W, Ncols, 0, n0, t);

  for (int i = 0; i < NK; ++i) {
    __syncthreads();
    commitA8(ra0, dA0);
    commitA8(ra1, dA1);
    commitB8(rb, Blds, t);
    __syncthreads();
    if (i + 1 < NK) {
      int kt = (i + 1) * BK;
      ra0 = fetchA8(pa0 + kt); ra1 = fetchA8(pa1 + kt);
      rb = fetchB8(W, Ncols, kt, n0, t);
    }
    FragAB a[2], b[2];
    load_fragA(Alds, wm * 32, lane, a[0]);
    load_fragA(Alds, wm * 32 + 16, lane, a[1]);
    load_fragB(Blds, wn * 32, lane, b[0]);
    load_fragB(Blds, wn * 32 + 16, lane, b[1]);
#pragma unroll
    for (int fm = 0; fm < 2; ++fm)
#pragma unroll
      for (int fn = 0; fn < 2; ++fn) acc[fm][fn] = WMMA_BF16(a[fm].v, b[fn].v, acc[fm][fn]);
  }

#pragma unroll
  for (int fm = 0; fm < 2; ++fm)
#pragma unroll
    for (int fn = 0; fn < 2; ++fn)
#pragma unroll
      for (int v = 0; v < 8; ++v) {
        int row = row0 + wm * 32 + fm * 16 + v + ((lane & 16) ? 8 : 0);
        int col = n0 + wn * 32 + fn * 16 + (lane & 15);
        if (row < M) {
          float x = acc[fm][fn][v] + bias[col];
          if (BF16OUT) {
            outBf[(size_t)row * Ncols + col] = f2bf(fmaxf(x, 0.f));
          } else {
            x += resid[(size_t)row * Ncols + col];
            if (do_relu) x = fmaxf(x, 0.f);
            outF[(size_t)row * Ncols + col] = x;
          }
        }
      }
}

// ---------------- small helper kernels ----------------
__global__ void cvt_f32_bf16_kernel(const float* __restrict__ s, unsigned short* __restrict__ d, int n) {
  int i = blockIdx.x * blockDim.x + threadIdx.x;
  if (i < n) d[i] = f2bf(s[i]);
}
__global__ void zero_f32_kernel(float* p, int n) {
  int i = blockIdx.x * blockDim.x + threadIdx.x;
  if (i < n) p[i] = 0.f;
}
__global__ void count_kernel(const int* __restrict__ dstIdx, float* __restrict__ cnt) {
  int i = blockIdx.x * blockDim.x + threadIdx.x;
  if (i < N_EDGES) atomicAdd(&cnt[dstIdx[i]], 1.0f);
}
__global__ void normalize_kernel(const float* __restrict__ agg, const float* __restrict__ cnt,
                                 unsigned short* __restrict__ aggb) {
  int i = blockIdx.x * blockDim.x + threadIdx.x;
  if (i < N_NODES * DH) {
    float c = fmaxf(cnt[i >> 9], 1.f);
    aggb[i] = f2bf(agg[i] / c);
  }
}

// ---------------- host launcher ----------------
extern "C" void kernel_launch(void* const* d_in, const int* in_sizes, int n_in,
                              void* d_out, int out_size, void* d_ws, size_t ws_size,
                              hipStream_t stream) {
  (void)in_sizes; (void)n_in; (void)out_size; (void)ws_size;
  const float* x_in = (const float*)d_in[0];
  const float* e_in = (const float*)d_in[1];
  const int* eidx   = (const int*)d_in[2];
  const int* srcIdx = eidx;
  const int* dstIdx = eidx + N_EDGES;
  const float* W1a = (const float*)d_in[3];  const float* b1a = (const float*)d_in[4];
  const float* W1b = (const float*)d_in[5];  const float* b1b = (const float*)d_in[6];
  const float* W2a = (const float*)d_in[7];  const float* b2a = (const float*)d_in[8];
  const float* W2b = (const float*)d_in[9];  const float* b2b = (const float*)d_in[10];

  // workspace carve
  char* ws = (char*)d_ws; size_t off = 0;
  auto carve = [&](size_t bytes) { void* p = ws + off; off += (bytes + 255) & ~(size_t)255; return p; };
  const size_t nW1a = (size_t)NLAYERS * (2 * DN + DE) * DH;
  const size_t nW1b = (size_t)NLAYERS * DH * (2 * DH + DE);
  const size_t nW2a = (size_t)NLAYERS * DH * DH;
  const size_t nW2b = (size_t)NLAYERS * DH * DN;
  unsigned short* wb1a = (unsigned short*)carve(nW1a * 2);
  unsigned short* wb1b = (unsigned short*)carve(nW1b * 2);
  unsigned short* wb2a = (unsigned short*)carve(nW2a * 2);
  unsigned short* wb2b = (unsigned short*)carve(nW2b * 2);
  unsigned short* xb   = (unsigned short*)carve((size_t)N_NODES * DN * 2);
  unsigned short* eb   = (unsigned short*)carve((size_t)N_EDGES * DE * 2);
  unsigned short* m1   = (unsigned short*)carve((size_t)N_EDGES * DH * 2);  // also aliases aggb/h
  float* agg  = (float*)carve((size_t)N_NODES * DH * 4);
  float* cnt  = (float*)carve((size_t)N_NODES * 4);
  float* x_ws = (float*)carve((size_t)N_NODES * DN * 4);
  float* e_ws = (float*)carve((size_t)N_EDGES * DE * 4);
  // alias: m1 (100MB) is dead after gemm2; reuse for aggb (20MB) + h (20MB)
  unsigned short* aggb = m1;
  unsigned short* h    = m1 + (size_t)N_NODES * DH;

  const int T = 256;
  auto g1 = [&](size_t n) { return (unsigned)((n + T - 1) / T); };

  // weights -> bf16 (every call; deterministic)
  cvt_f32_bf16_kernel<<<g1(nW1a), T, 0, stream>>>(W1a, wb1a, (int)nW1a);
  cvt_f32_bf16_kernel<<<g1(nW1b), T, 0, stream>>>(W1b, wb1b, (int)nW1b);
  cvt_f32_bf16_kernel<<<g1(nW2a), T, 0, stream>>>(W2a, wb2a, (int)nW2a);
  cvt_f32_bf16_kernel<<<g1(nW2b), T, 0, stream>>>(W2b, wb2b, (int)nW2b);

  // in-degree counts (dst is constant across layers)
  zero_f32_kernel<<<g1(N_NODES), T, 0, stream>>>(cnt, N_NODES);
  count_kernel<<<g1(N_EDGES), T, 0, stream>>>(dstIdx, cnt);

  const unsigned edgeTiles = (N_EDGES + BM - 1) / BM;  // 782
  const unsigned nodeTiles = (N_NODES + BM - 1) / BM;  // 157

  const float* xcur = x_in;
  const float* ecur = e_in;
  float* x_next[NLAYERS] = {x_ws, (float*)d_out};
  float* e_next[NLAYERS] = {e_ws, (float*)d_out + (size_t)N_NODES * DN};

  for (int l = 0; l < NLAYERS; ++l) {
    cvt_f32_bf16_kernel<<<g1((size_t)N_NODES * DN), T, 0, stream>>>(xcur, xb, N_NODES * DN);
    cvt_f32_bf16_kernel<<<g1((size_t)N_EDGES * DE), T, 0, stream>>>(ecur, eb, N_EDGES * DE);

    gemm1_gather_kernel<<<dim3(edgeTiles, DH / BN), T, 0, stream>>>(
        xb, eb, srcIdx, dstIdx, wb1a + (size_t)l * (2 * DN + DE) * DH, b1a + (size_t)l * DH, m1);

    zero_f32_kernel<<<g1((size_t)N_NODES * DH), T, 0, stream>>>(agg, N_NODES * DH);
    gemm2_fused_kernel<<<dim3(edgeTiles, DH / BN + DE / BN), T, 0, stream>>>(
        m1, wb1b + (size_t)l * DH * (2 * DH + DE), b1b + (size_t)l * (2 * DH + DE), dstIdx, agg,
        e_next[l]);

    normalize_kernel<<<g1((size_t)N_NODES * DH), T, 0, stream>>>(agg, cnt, aggb);

    gemm_plain_kernel<true><<<dim3(nodeTiles, DH / BN), T, 0, stream>>>(
        aggb, N_NODES, DH, DH, wb2a + (size_t)l * DH * DH, b2a + (size_t)l * DH, h, nullptr,
        nullptr, 0);

    gemm_plain_kernel<false><<<dim3(nodeTiles, DN / BN), T, 0, stream>>>(
        h, N_NODES, DH, DN, wb2b + (size_t)l * DH * DN, b2b + (size_t)l * DN, nullptr, x_next[l],
        xcur, (l < NLAYERS - 1) ? 1 : 0);

    xcur = x_next[l];
    ecur = e_next[l];
  }
}
